// Block_65987877535901
// MI455X (gfx1250) — matrix operand order repaired and checked by reading, not compiled, not run on hardware
//
#include <hip/hip_runtime.h>

// ---------------------------------------------------------------------------
// GPT block for MI455X (gfx1250): bf16 WMMA GEMMs + flash attention, f32 math
// elsewhere. Wave32. Global->LDS tiles that need no transpose are moved by the
// Tensor Data Mover (tensor_load_to_lds + s_wait_tensorcnt); TDM's LDS padding
// reproduces the padded strides the WMMA fragment loads expect.
// ---------------------------------------------------------------------------

typedef __bf16 bf16_t;
typedef __attribute__((ext_vector_type(8)))  bf16_t v8bf;
typedef __attribute__((ext_vector_type(16))) bf16_t v16bf;
typedef __attribute__((ext_vector_type(8)))  float  v8f;
typedef __attribute__((ext_vector_type(4)))  unsigned int u32x4;
typedef __attribute__((ext_vector_type(8)))  int          i32x8;
typedef __attribute__((ext_vector_type(4)))  int          i32x4;

#define N_EMBD   1024
#define T_SEQ    2048
#define BATCH    2
#define ROWS     (BATCH * T_SEQ)   // 4096 token rows
#define N_HEADS  16
#define HEAD_DIM 64

__device__ __forceinline__ v16bf cat8(v8bf lo, v8bf hi) {
  return __builtin_shufflevector(lo, hi, 0,1,2,3,4,5,6,7,8,9,10,11,12,13,14,15);
}
__device__ __forceinline__ v8f wmma_bf16(v16bf a, v16bf b, v8f c) {
  // D(f32 16x16) = A(16x32 bf16) * B(32x16 bf16) + C
  return __builtin_amdgcn_wmma_f32_16x16x32_bf16(false, a, false, b, (short)0, c,
                                                 false, false);
}

// ---------------------------------------------------------------------------
// TDM: 2D bf16 tile (tile_d0 x tile_d1, row stride stride_elems in memory)
// -> LDS at lds_addr, with per-row LDS padding.
// pad_interval_code: DWORDs between pads = 2<<code; pad_amount_code: pad = code+1 DWORDs.
// D# packing per CDNA5 ISA (2D: groups 2/3 zero).
// This toolchain's builtin is the 6-arg form:
//   (uint32x4 g0, int32x8 g1, int32x4, int32x4, int32x8, i32 cpol)
// ---------------------------------------------------------------------------
__device__ __forceinline__ void tdm_load_2d(unsigned int lds_addr,
                                            const void* gptr,
                                            unsigned int tile_d0,
                                            unsigned int tile_d1,
                                            unsigned long long stride_elems,
                                            unsigned int pad_interval_code,
                                            unsigned int pad_amount_code) {
  const unsigned long long ga = (unsigned long long)(uintptr_t)gptr;
  const unsigned int td0 = 0x3FFFFFFFu, td1 = 0x3FFFFFFFu;  // no OOB clipping
  u32x4 g0;
  g0[0] = 1u;                                   // count=1, user mode
  g0[1] = lds_addr;                             // lds_addr [63:32]
  g0[2] = (unsigned int)ga;                     // global_addr lo
  g0[3] = (unsigned int)((ga >> 32) & 0x1FFFFFFu) | (2u << 30);  // addr hi | type=2
  i32x8 g1;
  g1[0] = (int)((1u << 16)                      // data_size=1 (2 bytes)
              | (1u << 20)                      // pad_enable
              | (pad_interval_code << 22)
              | (pad_amount_code << 25));
  g1[1] = (int)((td0 & 0xFFFFu) << 16);         // [47:32]=0, [63:48]=td0 lo16
  g1[2] = (int)((td0 >> 16) | ((td1 & 0xFFFFu) << 16));
  g1[3] = (int)((td1 >> 16) | (tile_d0 << 16)); // tile_dim0 [127:112]
  g1[4] = (int)(tile_d1 & 0xFFFFu);             // tile_dim1; tile_dim2=0
  g1[5] = (int)(unsigned int)stride_elems;      // dim0_stride lo32
  g1[6] = (int)(unsigned int)((stride_elems >> 32) & 0xFFFFu);  // hi16; dim1_stride=0
  g1[7] = 0;
  i32x4 z4 = {0, 0, 0, 0};
  i32x8 z8 = {0, 0, 0, 0, 0, 0, 0, 0};
  __builtin_amdgcn_tensor_load_to_lds(g0, g1, z4, z4, z8, 0);
}

// ---------------------------------------------------------------------------
// f32 -> bf16 elementwise (weights)
// ---------------------------------------------------------------------------
__global__ void f32_to_bf16(const float* __restrict__ in,
                            bf16_t* __restrict__ out, int n) {
  int i = blockIdx.x * 256 + threadIdx.x;
  if (i < n) out[i] = (bf16_t)in[i];
}

// ---------------------------------------------------------------------------
// LayerNorm (f32 in, bf16 out). One 256-thread block per 1024-wide row.
// ---------------------------------------------------------------------------
__global__ __launch_bounds__(256) void layernorm_to_bf16(
    const float* __restrict__ x, const float* __restrict__ g,
    const float* __restrict__ b, bf16_t* __restrict__ out) {
  __shared__ float s1[256], s2[256];
  const int row = blockIdx.x, tid = threadIdx.x;
  const float* xr = x + (size_t)row * N_EMBD;
  float4 v = *(const float4*)(xr + tid * 4);
  s1[tid] = v.x + v.y + v.z + v.w;
  s2[tid] = v.x * v.x + v.y * v.y + v.z * v.z + v.w * v.w;
  __syncthreads();
  for (int s = 128; s > 0; s >>= 1) {
    if (tid < s) { s1[tid] += s1[tid + s]; s2[tid] += s2[tid + s]; }
    __syncthreads();
  }
  float mean = s1[0] * (1.0f / N_EMBD);
  float var  = s2[0] * (1.0f / N_EMBD) - mean * mean;
  float inv  = rsqrtf(var + 1e-5f);
  float4 gg = *(const float4*)(g + tid * 4);
  float4 bb = *(const float4*)(b + tid * 4);
  bf16_t* o = out + (size_t)row * N_EMBD + tid * 4;
  o[0] = (bf16_t)((v.x - mean) * inv * gg.x + bb.x);
  o[1] = (bf16_t)((v.y - mean) * inv * gg.y + bb.y);
  o[2] = (bf16_t)((v.z - mean) * inv * gg.z + bb.z);
  o[3] = (bf16_t)((v.w - mean) * inv * gg.w + bb.w);
}

// ---------------------------------------------------------------------------
// Tiled bf16 WMMA GEMM: C[M,N] = A[M,K] * B[K,N] (+bias, epilogue).
// Block tile 128x128, 8 waves (4x2), wave tile 32x64 (2x4 WMMA accumulators).
// A tile: TDM load (HW-padded to LDT stride). B tile: manual transpose store.
// EPI: 0 = bias -> bf16 out
//      1 = bias + exact GELU -> bf16 out
//      2 = bias + residual  -> f32 out
// ---------------------------------------------------------------------------
#define LDT 40  // padded LDS K-stride (elements); 80B rows keep b128 alignment

template <int EPI>
__global__ __launch_bounds__(256) void gemm_bf16_wmma(
    const bf16_t* __restrict__ A, const bf16_t* __restrict__ Bm,
    const float* __restrict__ bias, const float* __restrict__ resid,
    void* __restrict__ Cout, int M, int N, int K) {
  __shared__ bf16_t sA[128 * LDT];   // A tile, row-major [m][k], stride LDT
  __shared__ bf16_t sB[128 * LDT];   // B tile transposed, [n][k]
  const int tid = threadIdx.x;
  const int wid = tid >> 5, lane = tid & 31;
  const int l16 = lane & 15, hi = lane >> 4;
  const int wm = wid >> 1, wn = wid & 1;
  const int m0 = blockIdx.y * 128, n0 = blockIdx.x * 128;

  const int bkr = tid >> 3, bnseg = (tid & 7) * 16;    // B: 32 rows x 128
  const unsigned int ldsA = (unsigned int)(uintptr_t)(void*)sA;

  v8f zero = {};
  v8f acc[2][4];
  for (int mt = 0; mt < 2; ++mt)
    for (int nt = 0; nt < 4; ++nt) acc[mt][nt] = zero;

  for (int k0 = 0; k0 < K; k0 += 32) {
    // ---- A tile via Tensor Data Mover: 32x128 tile, rows padded 16+4 DWORDs ----
    if (wid == 0) {
      tdm_load_2d(ldsA, A + (size_t)m0 * K + k0,
                  /*tile_d0=*/32, /*tile_d1=*/128,
                  /*stride=*/(unsigned long long)K,
                  /*interval 16 DW=*/3u, /*pad 4 DW=*/3u);
    }
    // ---- B tile: global -> LDS transposed ----
    const bf16_t* gb = Bm + (size_t)(k0 + bkr) * N + n0 + bnseg;
    v8bf b0 = *(const v8bf*)gb;
    v8bf b1 = *(const v8bf*)(gb + 8);
#pragma unroll
    for (int i = 0; i < 8; ++i) {           // transpose on store
      sB[(bnseg + i)     * LDT + bkr] = b0[i];
      sB[(bnseg + 8 + i) * LDT + bkr] = b1[i];
    }
    if (k0 + 32 < K)                        // gfx1250 global_prefetch_b8
      __builtin_prefetch(Bm + (size_t)(k0 + 32 + bkr) * N + n0 + bnseg, 0, 0);
    if (wid == 0) __builtin_amdgcn_s_wait_tensorcnt(0);
    __syncthreads();

    // ---- LDS -> fragments (ISA VGPR layouts) ----
    v16bf af[2], bfm[4];
#pragma unroll
    for (int mt = 0; mt < 2; ++mt) {
      int base = (wm * 32 + mt * 16 + l16) * LDT + hi * 8;   // A: K {0..7|8..15} then +16
      af[mt] = cat8(*(const v8bf*)&sA[base], *(const v8bf*)&sA[base + 16]);
    }
#pragma unroll
    for (int nt = 0; nt < 4; ++nt) {
      int base = (wn * 64 + nt * 16 + l16) * LDT + hi * 8;   // B: K {0..7|8..15} then +16
      bfm[nt] = cat8(*(const v8bf*)&sB[base], *(const v8bf*)&sB[base + 16]);
    }
#pragma unroll
    for (int mt = 0; mt < 2; ++mt)
#pragma unroll
      for (int nt = 0; nt < 4; ++nt)
        acc[mt][nt] = wmma_bf16(af[mt], bfm[nt], acc[mt][nt]);
    __syncthreads();
  }

  // ---- epilogue (C layout: row = j + 8*hi, col = l16) ----
#pragma unroll
  for (int mt = 0; mt < 2; ++mt) {
#pragma unroll
    for (int nt = 0; nt < 4; ++nt) {
      const int cn = n0 + wn * 64 + nt * 16 + l16;
      const float bv = bias[cn];
#pragma unroll
      for (int j = 0; j < 8; ++j) {
        const int r = m0 + wm * 32 + mt * 16 + j + 8 * hi;
        float v = acc[mt][nt][j] + bv;
        if (EPI == 1) v = 0.5f * v * (1.0f + erff(v * 0.70710678118654752f));
        if (EPI == 2) {
          ((float*)Cout)[(size_t)r * N + cn] = v + resid[(size_t)r * N + cn];
        } else {
          ((bf16_t*)Cout)[(size_t)r * N + cn] = (bf16_t)v;
        }
      }
    }
  }
}

// ---------------------------------------------------------------------------
// Flash attention: block = (64 queries) x (one batch-head), 4 waves,
// 16 query rows per wave, key blocks of 32, online softmax in f32.
// K tile via TDM (HW-padded stride 80); V transposed manually; Q direct frags.
// qkv: [ROWS, 3*N_EMBD] bf16 (q|k|v).  y: [ROWS, N_EMBD] bf16.
// ---------------------------------------------------------------------------
__global__ __launch_bounds__(128) void attention_flash(
    const bf16_t* __restrict__ qkv, bf16_t* __restrict__ y) {
  __shared__ bf16_t sK[32 * 80];        // K block row-major [key][d], pad 80
  __shared__ bf16_t sV[64 * 40];        // V block transposed [d][key], pad 40
  __shared__ bf16_t sP[4][16 * 40];     // per-wave P scratch [row][key], pad 40

  const int bh = blockIdx.y;
  const int bidx = bh >> 4, h = bh & 15;
  const int qb0 = blockIdx.x * 64;
  const int tid = threadIdx.x;
  const int wid = tid >> 5, lane = tid & 31;
  const int l16 = lane & 15, hi = lane >> 4;
  const bf16_t* base = qkv + (size_t)bidx * T_SEQ * (3 * N_EMBD);
  const unsigned int ldsK = (unsigned int)(uintptr_t)(void*)sK;

  // Q fragments for this wave's 16 rows (A layout), scale applied to scores.
  v16bf qf[2];
  {
    const bf16_t* qp =
        base + (size_t)(qb0 + wid * 16 + l16) * (3 * N_EMBD) + h * HEAD_DIM;
#pragma unroll
    for (int c = 0; c < 2; ++c) {
      int off = c * 32 + hi * 8;
      qf[c] = cat8(*(const v8bf*)(qp + off), *(const v8bf*)(qp + off + 16));
    }
  }

  v8f zero = {};
  v8f o[4];
  for (int dt = 0; dt < 4; ++dt) o[dt] = zero;
  float mrow[8], lrow[8];
#pragma unroll
  for (int j = 0; j < 8; ++j) { mrow[j] = -1e30f; lrow[j] = 0.0f; }

  const int nkb = (qb0 >> 5) + 2;            // causal: keys up to qb0+63
  const int key = tid >> 2, dseg = (tid & 3) * 16;

  for (int kb = 0; kb < nkb; ++kb) {
    const int k0 = kb * 32;
    // K block via TDM: 64x32 tile, rows padded 32+8 DWORDs -> stride 80 elems
    if (wid == 0) {
      tdm_load_2d(ldsK,
                  base + (size_t)k0 * (3 * N_EMBD) + N_EMBD + h * HEAD_DIM,
                  /*tile_d0=*/64, /*tile_d1=*/32,
                  /*stride=*/(unsigned long long)(3 * N_EMBD),
                  /*interval 32 DW=*/4u, /*pad 8 DW=*/7u);
    }
    // V block -> sV transposed (B for P@V)
    const bf16_t* vp =
        base + (size_t)(k0 + key) * (3 * N_EMBD) + 2 * N_EMBD + h * HEAD_DIM + dseg;
    v8bf va = *(const v8bf*)vp, vb = *(const v8bf*)(vp + 8);
#pragma unroll
    for (int i = 0; i < 8; ++i) {
      sV[(dseg + i)     * 40 + key] = va[i];
      sV[(dseg + 8 + i) * 40 + key] = vb[i];
    }
    if (wid == 0) __builtin_amdgcn_s_wait_tensorcnt(0);
    __syncthreads();

    // S[16 x 32] = Q @ K^T  (two 16x16 C tiles, K-dim = 64 in 2 chunks)
    v8f s[2];
    s[0] = zero; s[1] = zero;
#pragma unroll
    for (int nt = 0; nt < 2; ++nt)
#pragma unroll
      for (int c = 0; c < 2; ++c) {
        int boff = (nt * 16 + l16) * 80 + c * 32 + hi * 8;
        v16bf kf = cat8(*(const v8bf*)&sK[boff], *(const v8bf*)&sK[boff + 16]);
        s[nt] = wmma_bf16(qf[c], kf, s[nt]);
      }

    // Online softmax per row (row = j + 8*hi, this lane holds cols l16, 16+l16)
    bf16_t* pw = &sP[wid][0];
#pragma unroll
    for (int j = 0; j < 8; ++j) {
      const int q = qb0 + wid * 16 + j + 8 * hi;
      float v0 = s[0][j] * 0.125f;          // 1/sqrt(64)
      float v1 = s[1][j] * 0.125f;
      if (k0 + l16 > q)      v0 = -1e30f;   // causal mask
      if (k0 + 16 + l16 > q) v1 = -1e30f;
      float mx = fmaxf(v0, v1);
      mx = fmaxf(mx, __shfl_xor(mx, 1, 16));
      mx = fmaxf(mx, __shfl_xor(mx, 2, 16));
      mx = fmaxf(mx, __shfl_xor(mx, 4, 16));
      mx = fmaxf(mx, __shfl_xor(mx, 8, 16));
      float mnew  = fmaxf(mrow[j], mx);
      float alpha = __expf(mrow[j] - mnew);
      mrow[j] = mnew;
      float p0 = __expf(v0 - mnew), p1 = __expf(v1 - mnew);
      float rs = p0 + p1;
      rs += __shfl_xor(rs, 1, 16);
      rs += __shfl_xor(rs, 2, 16);
      rs += __shfl_xor(rs, 4, 16);
      rs += __shfl_xor(rs, 8, 16);
      lrow[j] = lrow[j] * alpha + rs;
#pragma unroll
      for (int dt = 0; dt < 4; ++dt) o[dt][j] = o[dt][j] * alpha;
      pw[(j + 8 * hi) * 40 + l16]      = (bf16_t)p0;   // C-layout -> LDS
      pw[(j + 8 * hi) * 40 + 16 + l16] = (bf16_t)p1;
    }

    // P (A-fragment, via per-wave LDS; DS ops are in-order within a wave)
    {
      int poff = l16 * 40 + hi * 8;
      v16bf pf = cat8(*(const v8bf*)&pw[poff], *(const v8bf*)&pw[poff + 16]);
#pragma unroll
      for (int dt = 0; dt < 4; ++dt) {
        int voff = (dt * 16 + l16) * 40 + hi * 8;
        v16bf vf = cat8(*(const v8bf*)&sV[voff], *(const v8bf*)&sV[voff + 16]);
        o[dt] = wmma_bf16(pf, vf, o[dt]);
      }
    }
    __syncthreads();
  }

  // Normalize and store (y is [ROWS, N_EMBD] bf16)
#pragma unroll
  for (int j = 0; j < 8; ++j) {
    const int q = qb0 + wid * 16 + j + 8 * hi;
    const float inv = 1.0f / lrow[j];
    bf16_t* yr = y + (size_t)q * N_EMBD + h * HEAD_DIM + l16;
#pragma unroll
    for (int dt = 0; dt < 4; ++dt) yr[dt * 16] = (bf16_t)(o[dt][j] * inv);
  }
}

// ---------------------------------------------------------------------------
// Launch: LN1 -> QKV -> attention -> proj(+x) -> LN2 -> fc(GELU) -> fc2(+x1)
// ---------------------------------------------------------------------------
extern "C" void kernel_launch(void* const* d_in, const int* in_sizes, int n_in,
                              void* d_out, int out_size, void* d_ws,
                              size_t ws_size, hipStream_t stream) {
  (void)in_sizes; (void)n_in; (void)out_size; (void)ws_size;
  const float* x      = (const float*)d_in[0];
  const float* ln1_g  = (const float*)d_in[1];
  const float* ln1_b  = (const float*)d_in[2];
  const float* W_attn = (const float*)d_in[3];
  const float* b_attn = (const float*)d_in[4];
  const float* W_proj = (const float*)d_in[5];
  const float* b_proj = (const float*)d_in[6];
  const float* ln2_g  = (const float*)d_in[7];
  const float* ln2_b  = (const float*)d_in[8];
  const float* W_fc   = (const float*)d_in[9];
  const float* b_fc   = (const float*)d_in[10];
  const float* W_fc2  = (const float*)d_in[11];
  const float* b_fc2  = (const float*)d_in[12];

  char* ws = (char*)d_ws;
  const size_t MiB = 1ull << 20;
  bf16_t* ln_out = (bf16_t*)(ws + 0);         //  8 MiB  [4096,1024] bf16
  bf16_t* qkvb   = (bf16_t*)(ws + 8 * MiB);   // 24 MiB  [4096,3072] bf16
  bf16_t* ybuf   = (bf16_t*)(ws + 32 * MiB);  //  8 MiB  [4096,1024] bf16
  bf16_t* hbuf   = (bf16_t*)(ws + 8 * MiB);   // 32 MiB  [4096,4096] bf16 (reuses qkv+y)
  float*  x1     = (float*)(ws + 40 * MiB);   // 16 MiB  [4096,1024] f32
  bf16_t* Wa     = (bf16_t*)(ws + 56 * MiB);  //  6 MiB
  bf16_t* Wp     = (bf16_t*)(ws + 62 * MiB);  //  2 MiB
  bf16_t* Wf     = (bf16_t*)(ws + 64 * MiB);  //  8 MiB
  bf16_t* Wf2    = (bf16_t*)(ws + 72 * MiB);  //  8 MiB -> 80 MiB total

  // Weight conversion (all sizes are multiples of 256)
  f32_to_bf16<<<(1024 * 3072) / 256, 256, 0, stream>>>(W_attn, Wa, 1024 * 3072);
  f32_to_bf16<<<(1024 * 1024) / 256, 256, 0, stream>>>(W_proj, Wp, 1024 * 1024);
  f32_to_bf16<<<(1024 * 4096) / 256, 256, 0, stream>>>(W_fc,  Wf,  1024 * 4096);
  f32_to_bf16<<<(4096 * 1024) / 256, 256, 0, stream>>>(W_fc2, Wf2, 4096 * 1024);

  layernorm_to_bf16<<<ROWS, 256, 0, stream>>>(x, ln1_g, ln1_b, ln_out);
  gemm_bf16_wmma<0><<<dim3(3072 / 128, ROWS / 128), 256, 0, stream>>>(
      ln_out, Wa, b_attn, nullptr, qkvb, ROWS, 3072, 1024);
  attention_flash<<<dim3(T_SEQ / 64, BATCH * N_HEADS), 128, 0, stream>>>(qkvb, ybuf);
  gemm_bf16_wmma<2><<<dim3(1024 / 128, ROWS / 128), 256, 0, stream>>>(
      ybuf, Wp, b_proj, x, x1, ROWS, 1024, 1024);
  layernorm_to_bf16<<<ROWS, 256, 0, stream>>>(x1, ln2_g, ln2_b, ln_out);
  gemm_bf16_wmma<1><<<dim3(4096 / 128, ROWS / 128), 256, 0, stream>>>(
      ln_out, Wf, b_fc, nullptr, hbuf, ROWS, 4096, 1024);
  gemm_bf16_wmma<2><<<dim3(1024 / 128, ROWS / 128), 256, 0, stream>>>(
      hbuf, Wf2, b_fc2, x1, (float*)d_out, ROWS, 1024, 4096);
}
